// MultiHeadAttentionLayer_62723702391167
// MI455X (gfx1250) — matrix operand order repaired and checked by reading, not compiled, not run on hardware
//
#include <hip/hip_runtime.h>
#include <hip/hip_bf16.h>
#include <cstdint>

// ---------------------------------------------------------------------------
// MHA for MI455X (gfx1250): bf16 WMMA (v_wmma_f32_16x16x32_bf16) everywhere,
// async global->LDS staging for the attention K tile, WMMA-based row sums.
// B=4, S=2048, D=1024, H=16, depth=64.
// ---------------------------------------------------------------------------

typedef __bf16 v16bf __attribute__((ext_vector_type(16)));
typedef float  v8f   __attribute__((ext_vector_type(8)));

#define MHA_B 4
#define MHA_S 2048
#define MHA_D 1024
#define MHA_H 16
#define MHA_DEPTH 64

__device__ __forceinline__ __bf16 f2bf(float f) { return (__bf16)f; }  // RNE cvt
__device__ __forceinline__ __bf16 to_bf(float f)  { return f2bf(f); }
__device__ __forceinline__ __bf16 to_bf(__bf16 x) { return x; }
__device__ __forceinline__ void store_val(float* p, float v)  { *p = v; }
__device__ __forceinline__ void store_val(__bf16* p, float v) { *p = f2bf(v); }

union BfPack { __bf16 h[8]; uint4 v; };

__device__ __forceinline__ v8f v8f_zero() {
  v8f z;
#pragma unroll
  for (int i = 0; i < 8; ++i) z[i] = 0.0f;
  return z;
}

// ---- CDNA5 async global->LDS (ASYNCcnt) ------------------------------------
// LDS address comes from the VDST VGPR operand; low 32 bits of a generic shared pointer are
// the LDS byte address (aperture base lives in addr[63:32]).
__device__ __forceinline__ void async_load_b128(const void* gptr, void* ldsptr) {
  unsigned lo = (unsigned)(uintptr_t)ldsptr;
  unsigned long long ga = (unsigned long long)(uintptr_t)gptr;
  asm volatile("global_load_async_to_lds_b128 %0, %1, off"
               :: "v"(lo), "v"(ga) : "memory");
}
__device__ __forceinline__ void wait_async0() {
  asm volatile("s_wait_asynccnt 0x0" ::: "memory");
}

// A-fragment (16x32 bf16): row = lane%16.
// lanes 0-15:  e0..15 = K0..15   (khalf = 0)
// lanes 16-31: e0..15 = K16..31  (khalf = 8)
__device__ __forceinline__ v16bf load_afrag(const __bf16* rowp, int khalf) {
  v16bf a;
#pragma unroll
  for (int j = 0; j < 8; ++j) {
    a[j]     = rowp[2 * khalf + j];
    a[8 + j] = rowp[2 * khalf + 8 + j];
  }
  return a;
}

// B-fragment (32x16 bf16): col = lane%16.
// lanes 0-15: e j = K j; lanes 16-31: e j = K 16+j. p = &tile[col][kbase].
__device__ __forceinline__ v16bf load_bfrag(const __bf16* p) {
  v16bf b;
#pragma unroll
  for (int j = 0; j < 16; ++j) b[j] = p[j];
  return b;
}

// ---------------------------------------------------------------------------
// GEMM + bias: C[M,N] = A[M,K] * W[K,N] + bias[N]
// Block tile 128x128, 8 waves (256 thr), wave tile 64x32, K-step 32.
// ---------------------------------------------------------------------------
#define TILE_M 128
#define TILE_N 128
#define TILE_K 32
#define LPAD 8   // pitch 40 elements = 80B (16B multiple)

// Stage 16 elements of one A row into LDS as bf16 (vectorized both ends).
__device__ __forceinline__ void stage_a_row(const float* src, __bf16* dst) {
  const float4* s4 = (const float4*)src;
#pragma unroll
  for (int g = 0; g < 2; ++g) {
    float4 x0 = s4[2 * g], x1 = s4[2 * g + 1];
    BfPack pk;
    pk.h[0] = f2bf(x0.x); pk.h[1] = f2bf(x0.y);
    pk.h[2] = f2bf(x0.z); pk.h[3] = f2bf(x0.w);
    pk.h[4] = f2bf(x1.x); pk.h[5] = f2bf(x1.y);
    pk.h[6] = f2bf(x1.z); pk.h[7] = f2bf(x1.w);
    *(uint4*)(dst + 8 * g) = pk.v;
  }
}
__device__ __forceinline__ void stage_a_row(const __bf16* src, __bf16* dst) {
#pragma unroll
  for (int g = 0; g < 2; ++g)
    *(uint4*)(dst + 8 * g) = ((const uint4*)src)[g];
}

template <typename AT, typename OT>
__global__ __launch_bounds__(256) void gemm_bias_kernel(
    const AT* __restrict__ A, const float* __restrict__ W,
    const float* __restrict__ bias, OT* __restrict__ C,
    int M, int N, int K) {
  __shared__ __bf16 As[TILE_M][TILE_K + LPAD];
  __shared__ __bf16 Wt[TILE_N][TILE_K + LPAD];   // W tile stored transposed

  const int tid  = threadIdx.x;
  const int lane = tid & 31;
  const int wave = tid >> 5;
  const int wm   = wave & 1;   // 0..1 -> M offset 64*wm
  const int wn   = wave >> 1;  // 0..3 -> N offset 32*wn
  const int m0   = blockIdx.y * TILE_M;
  const int n0   = blockIdx.x * TILE_N;

  const int khalf = (lane < 16) ? 0 : 8;
  const int kbase = (lane < 16) ? 0 : 16;

  v8f acc[4][2];
#pragma unroll
  for (int mt = 0; mt < 4; ++mt)
#pragma unroll
    for (int nt = 0; nt < 2; ++nt) acc[mt][nt] = v8f_zero();

  for (int k0 = 0; k0 < K; k0 += TILE_K) {
    // stage A tile (128x32) -> bf16
    {
      const int row = tid >> 1;
      const int c0  = (tid & 1) * 16;
      stage_a_row(A + (size_t)(m0 + row) * K + k0 + c0, &As[row][c0]);
    }
    // stage W tile transposed (32x128 -> Wt[n][k]) -> bf16
    {
      const int kk  = tid >> 3;
      const int nn0 = (tid & 7) * 16;
      const float4* s4 = (const float4*)(W + (size_t)(k0 + kk) * N + n0 + nn0);
#pragma unroll
      for (int g = 0; g < 4; ++g) {
        float4 x = s4[g];
        Wt[nn0 + 4 * g + 0][kk] = f2bf(x.x);
        Wt[nn0 + 4 * g + 1][kk] = f2bf(x.y);
        Wt[nn0 + 4 * g + 2][kk] = f2bf(x.z);
        Wt[nn0 + 4 * g + 3][kk] = f2bf(x.w);
      }
    }
    // prefetch next K tile while this one is consumed (global_prefetch_b8)
    if (k0 + TILE_K < K) {
      __builtin_prefetch(A + (size_t)(m0 + (tid >> 1)) * K + k0 + TILE_K, 0, 1);
      __builtin_prefetch(W + (size_t)(k0 + TILE_K + (tid >> 3)) * N + n0, 0, 1);
    }
    __syncthreads();

    v16bf a[4], b[2];
#pragma unroll
    for (int mt = 0; mt < 4; ++mt)
      a[mt] = load_afrag(&As[wm * 64 + mt * 16 + (lane & 15)][0], khalf);
#pragma unroll
    for (int nt = 0; nt < 2; ++nt)
      b[nt] = load_bfrag(&Wt[wn * 32 + nt * 16 + (lane & 15)][kbase]);

#pragma unroll
    for (int mt = 0; mt < 4; ++mt)
#pragma unroll
      for (int nt = 0; nt < 2; ++nt)
        acc[mt][nt] = __builtin_amdgcn_wmma_f32_16x16x32_bf16(
            false, a[mt], false, b[nt], (short)0, acc[mt][nt], false, false);

    __syncthreads();
  }

  // epilogue: C-layout (vgpr r -> row r / r+8 by half; lane%16 -> col)
#pragma unroll
  for (int mt = 0; mt < 4; ++mt) {
#pragma unroll
    for (int nt = 0; nt < 2; ++nt) {
      const int col = n0 + wn * 32 + nt * 16 + (lane & 15);
      const float bb = bias[col];
#pragma unroll
      for (int r = 0; r < 8; ++r) {
        const int row = m0 + wm * 64 + mt * 16 + ((lane < 16) ? r : r + 8);
        store_val(&C[(size_t)row * N + col], acc[mt][nt][r] + bb);
      }
    }
  }
}

// ---------------------------------------------------------------------------
// Flash attention: one block per (64 q-rows, head, batch); 4 waves x 16 rows.
// K tile staged via async global->LDS; V tile transposed via VGPRs; row sums
// via an extra WMMA against an all-ones B fragment.
// ---------------------------------------------------------------------------
#define ATT_KT 32

__global__ __launch_bounds__(128) void attention_kernel(
    const __bf16* __restrict__ Q, const __bf16* __restrict__ Km,
    const __bf16* __restrict__ V, const float* __restrict__ mask,
    __bf16* __restrict__ O) {
  __shared__ __bf16 Ks[ATT_KT][MHA_DEPTH + LPAD];   // [32][72] (keys x d)
  __shared__ __bf16 Vt[MHA_DEPTH][ATT_KT + LPAD];   // [64][40] (d x keys)
  __shared__ __bf16 Ps[4][16][ATT_KT + LPAD];       // per-wave P tile

  const int tid  = threadIdx.x;
  const int lane = tid & 31;
  const int wave = tid >> 5;
  const int b = blockIdx.z, h = blockIdx.y;
  const int q0 = blockIdx.x * 64 + wave * 16;
  const size_t headoff = (size_t)h * MHA_DEPTH;

  const int khalf = (lane < 16) ? 0 : 8;
  const int kbase = (lane < 16) ? 0 : 16;

  // Q fragments for this wave's 16 rows, d split into two K=32 chunks
  v16bf qa[2];
  {
    const __bf16* qrow =
        Q + ((size_t)b * MHA_S + q0 + (lane & 15)) * MHA_D + headoff;
#pragma unroll
    for (int c = 0; c < 2; ++c) qa[c] = load_afrag(qrow + c * 32, khalf);
  }

  // all-ones B fragment for row-sum WMMA
  v16bf ones;
  {
    const __bf16 onebf = __builtin_bit_cast(__bf16, (unsigned short)0x3F80);
#pragma unroll
    for (int j = 0; j < 16; ++j) ones[j] = onebf;
  }

  v8f accO[4];
#pragma unroll
  for (int nt = 0; nt < 4; ++nt) accO[nt] = v8f_zero();
  float mrun[8], lrun[8], alphas[8];
#pragma unroll
  for (int r = 0; r < 8; ++r) { mrun[r] = -1e30f; lrun[r] = 0.0f; }

  const float scale = 0.125f;  // 1/sqrt(64)
  const int nkt = MHA_S / ATT_KT;

  for (int kt = 0; kt < nkt; ++kt) {
    const int key0 = kt * ATT_KT;
    // --- stage K tile via async global->LDS (2 x b128 per thread) ----------
    {
      const int r  = tid >> 2;          // 0..31 (key within tile)
      const int c0 = (tid & 3) * 16;    // d offset
      const __bf16* ksrc =
          Km + ((size_t)b * MHA_S + key0 + r) * MHA_D + headoff + c0;
      async_load_b128(ksrc,     &Ks[r][c0]);
      async_load_b128(ksrc + 8, &Ks[r][c0 + 8]);

      // --- stage V tile transposed via VGPRs -------------------------------
      const __bf16* vsrc =
          V + ((size_t)b * MHA_S + key0 + r) * MHA_D + headoff + c0;
      BfPack a0, a1;
      a0.v = ((const uint4*)vsrc)[0];
      a1.v = ((const uint4*)vsrc)[1];
#pragma unroll
      for (int j = 0; j < 8; ++j) {
        Vt[c0 + j][r]     = a0.h[j];
        Vt[c0 + 8 + j][r] = a1.h[j];
      }
      if (kt + 1 < nkt)
        __builtin_prefetch(vsrc + (size_t)ATT_KT * MHA_D, 0, 0);
    }
    wait_async0();
    __syncthreads();

    // logits S = Q * K^T : two 16x16 key tiles, d accumulated over 2 chunks
    v8f sacc[2];
#pragma unroll
    for (int nt = 0; nt < 2; ++nt) {
      sacc[nt] = v8f_zero();
#pragma unroll
      for (int c = 0; c < 2; ++c) {
        v16bf bk = load_bfrag(&Ks[nt * 16 + (lane & 15)][c * 32 + kbase]);
        sacc[nt] = __builtin_amdgcn_wmma_f32_16x16x32_bf16(
            false, qa[c], false, bk, (short)0, sacc[nt], false, false);
      }
    }

    // additive mask, per key column
    const float madd0 = mask[(size_t)b * MHA_S + key0 + (lane & 15)] * -1e9f;
    const float madd1 = mask[(size_t)b * MHA_S + key0 + 16 + (lane & 15)] * -1e9f;

    // online softmax per row slot (rows r / r+8 by lane half); max via shfl
#pragma unroll
    for (int r = 0; r < 8; ++r) {
      float s0 = sacc[0][r] * scale + madd0;
      float s1 = sacc[1][r] * scale + madd1;
      float mx = fmaxf(s0, s1);
#pragma unroll
      for (int off = 8; off >= 1; off >>= 1)
        mx = fmaxf(mx, __shfl_xor(mx, off, 32));   // stays within 16-lane half
      const float mnew = fmaxf(mrun[r], mx);
      alphas[r] = __expf(mrun[r] - mnew);
      mrun[r] = mnew;
      const float p0 = __expf(s0 - mnew);
      const float p1 = __expf(s1 - mnew);
#pragma unroll
      for (int nt = 0; nt < 4; ++nt) accO[nt][r] *= alphas[r];
      const int prow = (lane < 16) ? r : r + 8;
      Ps[wave][prow][(lane & 15)]      = f2bf(p0);
      Ps[wave][prow][16 + (lane & 15)] = f2bf(p1);
    }

    // P fragment from per-wave LDS tile (A layout)
    v16bf pa = load_afrag(&Ps[wave][lane & 15][0], khalf);

    // row sums via WMMA against ones: every lane gets rowsum[its rows]
    v8f rowsum = __builtin_amdgcn_wmma_f32_16x16x32_bf16(
        false, pa, false, ones, (short)0, v8f_zero(), false, false);
#pragma unroll
    for (int r = 0; r < 8; ++r) lrun[r] = lrun[r] * alphas[r] + rowsum[r];

    // O += P * V
#pragma unroll
    for (int nt = 0; nt < 4; ++nt) {
      v16bf bvv = load_bfrag(&Vt[nt * 16 + (lane & 15)][kbase]);
      accO[nt] = __builtin_amdgcn_wmma_f32_16x16x32_bf16(
          false, pa, false, bvv, (short)0, accO[nt], false, false);
    }
    __syncthreads();
  }

  // final normalize + store context (bf16), layout [B,S,H*depth]
#pragma unroll
  for (int nt = 0; nt < 4; ++nt) {
    const int col = nt * 16 + (lane & 15);
#pragma unroll
    for (int r = 0; r < 8; ++r) {
      const int row = q0 + ((lane < 16) ? r : r + 8);
      const float ov = accO[nt][r] / lrun[r];
      O[((size_t)b * MHA_S + row) * MHA_D + headoff + col] = f2bf(ov);
    }
  }
}

// ---------------------------------------------------------------------------
extern "C" void kernel_launch(void* const* d_in, const int* in_sizes, int n_in,
                              void* d_out, int out_size, void* d_ws, size_t ws_size,
                              hipStream_t stream) {
  (void)in_sizes; (void)n_in; (void)out_size; (void)ws_size;
  const float* query = (const float*)d_in[0];
  const float* key   = (const float*)d_in[1];
  const float* value = (const float*)d_in[2];
  const float* mask  = (const float*)d_in[3];
  const float* Wq = (const float*)d_in[4];
  const float* bq = (const float*)d_in[5];
  const float* Wk = (const float*)d_in[6];
  const float* bk = (const float*)d_in[7];
  const float* Wv = (const float*)d_in[8];
  const float* bv = (const float*)d_in[9];
  const float* Wo = (const float*)d_in[10];
  const float* bo = (const float*)d_in[11];

  const int M = MHA_B * MHA_S;  // 8192
  const int N = MHA_D;          // 1024
  const int K = MHA_D;          // 1024
  const size_t elems = (size_t)M * MHA_D;

  char* ws = (char*)d_ws;
  __bf16* Qb = (__bf16*)(ws);
  __bf16* Kb = (__bf16*)(ws + elems * 2);
  __bf16* Vb = (__bf16*)(ws + elems * 4);
  __bf16* Cx = (__bf16*)(ws + elems * 6);

  dim3 gg(N / TILE_N, M / TILE_M);
  gemm_bias_kernel<float, __bf16><<<gg, 256, 0, stream>>>(query, Wq, bq, Qb, M, N, K);
  gemm_bias_kernel<float, __bf16><<<gg, 256, 0, stream>>>(key,   Wk, bk, Kb, M, N, K);
  gemm_bias_kernel<float, __bf16><<<gg, 256, 0, stream>>>(value, Wv, bv, Vb, M, N, K);

  dim3 ga(MHA_S / 64, MHA_H, MHA_B);
  attention_kernel<<<ga, 128, 0, stream>>>(Qb, Kb, Vb, mask, Cx);

  gemm_bias_kernel<__bf16, float><<<gg, 256, 0, stream>>>(Cx, Wo, bo, (float*)d_out, M, N, K);
}
